// RSSMCore_13812614824428
// MI455X (gfx1250) — matrix operand order repaired and checked by reading, not compiled, not run on hardware
//
#include <hip/hip_runtime.h>
#include <hip/hip_bf16.h>
#include <math.h>
#include <stdint.h>

// ---------------------------------------------------------------------------
// RSSM scan for MI455X (gfx1250): bf16 WMMA GEMMs, f32 accumulation,
// double-buffered LDS, async global->LDS fills (ASYNCcnt) when available.
// ---------------------------------------------------------------------------

typedef __bf16 bf16;
typedef __attribute__((ext_vector_type(16))) __bf16 v16bf;
typedef __attribute__((ext_vector_type(8)))  __bf16 v8bf;
typedef __attribute__((ext_vector_type(8)))  float  v8f;
typedef __attribute__((__vector_size__(4 * sizeof(int)))) int v4i;

#define T_STEPS 64
#define BATCH   512
#define E_DIM   1536
#define A_DIM   7
#define D_DIM   2048
#define S_DIM   32
#define H_DIM   1024
#define G_DIM   (3*D_DIM)      // 6144
#define DS_DIM  (D_DIM+S_DIM)  // 2080
#define DE_DIM  (D_DIM+E_DIM)  // 3584

#if defined(__AMDGCN__) && __has_builtin(__builtin_amdgcn_global_load_async_to_lds_b128) && \
    __has_builtin(__builtin_amdgcn_s_wait_asynccnt)
#define USE_ASYNC_LDS 1
#else
#define USE_ASYNC_LDS 0
#endif

#define AS3 __attribute__((address_space(3)))
#define AS1 __attribute__((address_space(1)))

__device__ __forceinline__ float elu_f(float x)      { return x > 0.f ? x : (__expf(x) - 1.f); }
__device__ __forceinline__ float sigm_f(float x)     { return 1.f / (1.f + __expf(-x)); }
__device__ __forceinline__ float softplus_f(float x) { return x > 20.f ? x : log1pf(__expf(x)); }

// ---------------------------- fp32 -> bf16 convert -------------------------
__global__ void cvt_f32_bf16(const float* __restrict__ src, bf16* __restrict__ dst, int n) {
    int i = blockIdx.x * blockDim.x + threadIdx.x;
    if (i < n) dst[i] = (bf16)src[i];
}

// ------------------- step begin: reset mask + h -> bf16 ---------------------
__global__ void step_begin(float* __restrict__ stateF32, bf16* __restrict__ hB,
                           const int* __restrict__ reset, int t) {
    int idx = blockIdx.x * blockDim.x + threadIdx.x;     // BATCH*DS_DIM
    if (idx >= BATCH * DS_DIM) return;
    int b = idx / DS_DIM, i = idx - b * DS_DIM;
    float s = reset[t * BATCH + b] ? 0.f : stateF32[idx];
    stateF32[idx] = s;
    if (i < D_DIM) hB[(size_t)b * D_DIM + i] = (bf16)s;
}

// -------------------- za = elu([z,a] @ za_W + za_b), K=39 -------------------
__global__ void za_kernel(const float* __restrict__ stateF32,
                          const float* __restrict__ action,
                          const float* __restrict__ za_W,
                          const float* __restrict__ za_b,
                          bf16* __restrict__ zaB, int t) {
    int idx = blockIdx.x * blockDim.x + threadIdx.x;     // BATCH*H_DIM
    if (idx >= BATCH * H_DIM) return;
    int b = idx >> 10, j = idx & (H_DIM - 1);
    float acc = za_b[j];
    const float* zrow = stateF32 + (size_t)b * DS_DIM + D_DIM;
    const float* arow = action + ((size_t)t * BATCH + b) * A_DIM;
#pragma unroll
    for (int k = 0; k < S_DIM; ++k) acc += zrow[k] * za_W[k * H_DIM + j];
#pragma unroll
    for (int k = 0; k < A_DIM; ++k) acc += arow[k] * za_W[(S_DIM + k) * H_DIM + j];
    zaB[idx] = (bf16)elu_f(acc);
}

// ------------------------- bf16 WMMA GEMM + bias ----------------------------
// C[M,N] = A[M,K](lda) @ B[K,N](ldb=N) + bias ; optional ELU ; f32 or bf16 out.
// WG tile 128xBN (8 waves: 4M x 2N; wave tile 32 x BN/2), K-step 32,
// double-buffered LDS, async A-tile fill when the builtin exists.
template<int BN, int ACT_ELU, int OUT_BF16>
__global__ __launch_bounds__(256)
void gemm_bf16_wmma(const bf16* __restrict__ A, int lda,
                    const bf16* __restrict__ B, int ldb,
                    const float* __restrict__ bias,
                    float* __restrict__ outF, bf16* __restrict__ outB,
                    int ldc, int K) {
    constexpr int NI    = BN / 32;      // per-wave 16-wide N subtiles (2 or 4)
    constexpr int BELEM = BN / 8;       // B elems per thread per K-tile (8 or 16)

    __shared__ __align__(32) bf16 ldsA [2][128 * 32];   // [m][k] row-major
    __shared__ __align__(32) bf16 ldsBt[2][BN * 32];    // [n][k] (transposed)

    const int tid  = threadIdx.x;
    const int lane = tid & 31, wave = tid >> 5;
    const int wm = wave >> 1, wn = wave & 1;            // wave grid 4 x 2
    const int blockM = blockIdx.y * 128;
    const int blockN = blockIdx.x * BN;
    const int l15 = lane & 15, lh = lane >> 4;

    // global-load assignments (256 threads)
    const int a_row = tid >> 1;            // 0..127
    const int a_col = (tid & 1) * 16;      // 0 / 16
    const int b_row = tid >> 3;            // 0..31 (k)
    const int b_col = (tid & 7) * BELEM;   // n offset

    const bf16* aptr = A + (size_t)(blockM + a_row) * lda + a_col;
    const bf16* bptr = B + (size_t)b_row * ldb + blockN + b_col;

    v8f acc[2][NI];
#pragma unroll
    for (int mi = 0; mi < 2; ++mi)
#pragma unroll
        for (int ni = 0; ni < NI; ++ni)
#pragma unroll
            for (int e = 0; e < 8; ++e) acc[mi][ni][e] = 0.f;

    auto loadBreg = [&](int kt, bf16* breg) {
        const bf16* bp = bptr + (size_t)kt * 32 * ldb;
#pragma unroll
        for (int c = 0; c < BELEM / 8; ++c)
            *(v8bf*)&breg[c * 8] = *(const v8bf*)(bp + c * 8);
    };
    auto storeBt = [&](const bf16* breg, int buf) {
#pragma unroll
        for (int e = 0; e < BELEM; ++e)
            ldsBt[buf][(b_col + e) * 32 + b_row] = breg[e];
    };
#if USE_ASYNC_LDS
    auto copyA = [&](int kt, int buf) {   // 16B/lane global -> LDS DMA (ASYNCcnt)
        __builtin_amdgcn_global_load_async_to_lds_b128(
            (AS1 v4i*)(uintptr_t)(aptr + (size_t)kt * 32),
            (AS3 v4i*)(unsigned)(uintptr_t)&ldsA[buf][a_row * 32 + a_col],
            0, 0);
    };
#endif

    auto compute = [&](int buf) {
        // A fragment: lane -> M=l15; contiguous k = lh*16..+15
        // (matches B).  B fragment: lane -> N=l15, contiguous
        // k = lh*16..+15 (B 32x16 layout); transposed LDS -> one 32B load.
        v16bf afrag[2], bfrag[NI];
#pragma unroll
        for (int mi = 0; mi < 2; ++mi) {
            int row = wm * 32 + mi * 16 + l15;
            afrag[mi] = *(const v16bf*)&ldsA[buf][row * 32 + lh * 16];
        }
#pragma unroll
        for (int ni = 0; ni < NI; ++ni) {
            int col = wn * (BN / 2) + ni * 16 + l15;
            bfrag[ni] = *(const v16bf*)&ldsBt[buf][col * 32 + lh * 16];
        }
#pragma unroll
        for (int mi = 0; mi < 2; ++mi)
#pragma unroll
            for (int ni = 0; ni < NI; ++ni)
                acc[mi][ni] = __builtin_amdgcn_wmma_f32_16x16x32_bf16(
                    false, afrag[mi], false, bfrag[ni],
                    (short)0, acc[mi][ni], false, false);
    };

    const int KT = K >> 5;

    // ---- prologue: stage K-tile 0 into buffer 0 ----
    {
        bf16 breg0[BELEM];
        loadBreg(0, breg0);
#if USE_ASYNC_LDS
        copyA(0, 0);
#else
        v16bf a0 = *(const v16bf*)aptr;
#endif
        storeBt(breg0, 0);
#if USE_ASYNC_LDS
        __builtin_amdgcn_s_wait_asynccnt(0);
#else
        *(v16bf*)&ldsA[0][a_row * 32 + a_col] = a0;
#endif
        __syncthreads();
    }

    // ---- main loop: fetch tile kt+1 while computing tile kt ----
    for (int kt = 0; kt < KT; ++kt) {
        const int cur = kt & 1, nxt = cur ^ 1;
        const bool have_next = (kt + 1 < KT);
        bf16 breg[BELEM];
#if !USE_ASYNC_LDS
        v16bf areg;
#endif
        if (have_next) {
            loadBreg(kt + 1, breg);
#if USE_ASYNC_LDS
            copyA(kt + 1, nxt);
#else
            areg = *(const v16bf*)(aptr + (size_t)(kt + 1) * 32);
#endif
            if (kt + 2 < KT)   // keep the tile after next warm in L2/WGP$
                __builtin_prefetch(bptr + (size_t)(kt + 2) * 32 * ldb, 0, 3);
        }

        compute(cur);

        if (have_next) {
            storeBt(breg, nxt);
#if USE_ASYNC_LDS
            __builtin_amdgcn_s_wait_asynccnt(0);
#else
            *(v16bf*)&ldsA[nxt][a_row * 32 + a_col] = areg;
#endif
            __syncthreads();
        }
    }

    // ---- epilogue: C layout -> m = base + lh*8 + v, n = base + l15 ----
#pragma unroll
    for (int mi = 0; mi < 2; ++mi)
#pragma unroll
        for (int ni = 0; ni < NI; ++ni)
#pragma unroll
            for (int v = 0; v < 8; ++v) {
                int m = blockM + wm * 32 + mi * 16 + lh * 8 + v;
                int n = blockN + wn * (BN / 2) + ni * 16 + l15;
                float val = acc[mi][ni][v] + bias[n];
                if (ACT_ELU) val = elu_f(val);
                if (OUT_BF16) outB[(size_t)m * ldc + n] = (bf16)val;
                else          outF[(size_t)m * ldc + n] = val;
            }
}

// --------------- GRU gating + [h_new, embed] bf16 concat --------------------
__global__ void gru_gate(const float* __restrict__ gi, const float* __restrict__ gh,
                         float* __restrict__ stateF32, bf16* __restrict__ heB,
                         const float* __restrict__ embed, int t) {
    int idx = blockIdx.x * blockDim.x + threadIdx.x;     // BATCH*DE_DIM
    if (idx >= BATCH * DE_DIM) return;
    int b = idx / DE_DIM, i = idx - b * DE_DIM;
    if (i < D_DIM) {
        float h  = stateF32[(size_t)b * DS_DIM + i];
        size_t g = (size_t)b * G_DIM + i;
        float rg = sigm_f(gi[g] + gh[g]);
        float zg = sigm_f(gi[g + D_DIM] + gh[g + D_DIM]);
        float n  = tanhf(gi[g + 2 * D_DIM] + rg * gh[g + 2 * D_DIM]);
        float hnew = (1.f - zg) * n + zg * h;
        stateF32[(size_t)b * DS_DIM + i] = hnew;
        heB[(size_t)b * DE_DIM + i] = (bf16)hnew;
    } else {
        int j = i - D_DIM;
        heB[(size_t)b * DE_DIM + i] = (bf16)embed[((size_t)t * BATCH + b) * E_DIM + j];
    }
}

// ------------- finalize: mean/std, sample, write outputs --------------------
__global__ void finalize(const float* __restrict__ priRaw, const float* __restrict__ posRaw,
                         const float* __restrict__ noise, float* __restrict__ stateF32,
                         float* __restrict__ priors, float* __restrict__ posts,
                         float* __restrict__ states, int t) {
    int idx = blockIdx.x * blockDim.x + threadIdx.x;     // BATCH*DS_DIM
    if (idx >= BATCH * DS_DIM) return;
    int b = idx / DS_DIM, i = idx - b * DS_DIM;
    size_t tb = (size_t)t * BATCH + b;
    if (i < D_DIM) {
        states[tb * DS_DIM + i] = stateF32[idx];
    } else {
        int j = i - D_DIM;                                // 0..31
        float pm = priRaw[(size_t)b * 2 * S_DIM + j];
        float ps = softplus_f(priRaw[(size_t)b * 2 * S_DIM + S_DIM + j]) + 0.1f;
        priors[tb * 2 * S_DIM + j] = pm;
        priors[tb * 2 * S_DIM + S_DIM + j] = ps;
        float qm = posRaw[(size_t)b * 2 * S_DIM + j];
        float qs = softplus_f(posRaw[(size_t)b * 2 * S_DIM + S_DIM + j]) + 0.1f;
        posts[tb * 2 * S_DIM + j] = qm;
        posts[tb * 2 * S_DIM + S_DIM + j] = qs;
        float samp = qm + qs * noise[tb * S_DIM + j];
        stateF32[(size_t)b * DS_DIM + D_DIM + j] = samp;
        states[tb * DS_DIM + D_DIM + j] = samp;
    }
}

// ---------------------------------------------------------------------------
extern "C" void kernel_launch(void* const* d_in, const int* in_sizes, int n_in,
                              void* d_out, int out_size, void* d_ws, size_t ws_size,
                              hipStream_t stream) {
    const float* embed    = (const float*)d_in[0];
    const float* action   = (const float*)d_in[1];
    const float* noise    = (const float*)d_in[2];
    const float* in_state = (const float*)d_in[3];
    const float* za_W     = (const float*)d_in[4];
    const float* za_b     = (const float*)d_in[5];
    const float* gru_Wi   = (const float*)d_in[6];
    const float* gru_Wh   = (const float*)d_in[7];
    const float* gru_bi   = (const float*)d_in[8];
    const float* gru_bh   = (const float*)d_in[9];
    const float* prior_W1 = (const float*)d_in[10];
    const float* prior_b1 = (const float*)d_in[11];
    const float* prior_W2 = (const float*)d_in[12];
    const float* prior_b2 = (const float*)d_in[13];
    const float* post_W1  = (const float*)d_in[14];
    const float* post_b1  = (const float*)d_in[15];
    const float* post_W2  = (const float*)d_in[16];
    const float* post_b2  = (const float*)d_in[17];
    const int*   reset    = (const int*)  d_in[18];

    float* priors = (float*)d_out;
    float* posts  = priors + (size_t)T_STEPS * BATCH * 2 * S_DIM;
    float* states = posts  + (size_t)T_STEPS * BATCH * 2 * S_DIM;

    // ---- workspace carve-out (256B aligned) ----
    char* ws = (char*)d_ws;
    size_t off = 0;
    auto carve = [&](size_t bytes) -> void* {
        void* p = ws + off;
        off = (off + bytes + 255) & ~(size_t)255;
        return p;
    };
    bf16*  WiB   = (bf16*) carve((size_t)H_DIM  * G_DIM  * 2);
    bf16*  WhB   = (bf16*) carve((size_t)D_DIM  * G_DIM  * 2);
    bf16*  P1B   = (bf16*) carve((size_t)D_DIM  * H_DIM  * 2);
    bf16*  P2B   = (bf16*) carve((size_t)H_DIM  * 2*S_DIM * 2);
    bf16*  Q1B   = (bf16*) carve((size_t)DE_DIM * H_DIM  * 2);
    bf16*  Q2B   = (bf16*) carve((size_t)H_DIM  * 2*S_DIM * 2);
    float* stateF= (float*)carve((size_t)BATCH * DS_DIM * 4);
    bf16*  hB    = (bf16*) carve((size_t)BATCH * D_DIM  * 2);
    bf16*  zaB   = (bf16*) carve((size_t)BATCH * H_DIM  * 2);
    float* giF   = (float*)carve((size_t)BATCH * G_DIM  * 4);
    float* ghF   = (float*)carve((size_t)BATCH * G_DIM  * 4);
    bf16*  heB   = (bf16*) carve((size_t)BATCH * DE_DIM * 2);
    bf16*  p1B   = (bf16*) carve((size_t)BATCH * H_DIM  * 2);
    bf16*  q1B   = (bf16*) carve((size_t)BATCH * H_DIM  * 2);
    float* priR  = (float*)carve((size_t)BATCH * 2*S_DIM * 4);
    float* posR  = (float*)carve((size_t)BATCH * 2*S_DIM * 4);

    // ---- one-time: weights -> bf16 (L2-resident thereafter) ----
    auto cvt = [&](const float* s, bf16* d, size_t n) {
        cvt_f32_bf16<<<dim3((unsigned)((n + 255) / 256)), 256, 0, stream>>>(s, d, (int)n);
    };
    cvt(gru_Wi,   WiB, (size_t)H_DIM  * G_DIM);
    cvt(gru_Wh,   WhB, (size_t)D_DIM  * G_DIM);
    cvt(prior_W1, P1B, (size_t)D_DIM  * H_DIM);
    cvt(prior_W2, P2B, (size_t)H_DIM  * 2 * S_DIM);
    cvt(post_W1,  Q1B, (size_t)DE_DIM * H_DIM);
    cvt(post_W2,  Q2B, (size_t)H_DIM  * 2 * S_DIM);

    (void)hipMemcpyAsync(stateF, in_state, (size_t)BATCH * DS_DIM * 4,
                         hipMemcpyDeviceToDevice, stream);

    auto gemm = [&](const bf16* Am, int lda, const bf16* Bw, const float* bias,
                    void* out, int N, int K, bool elu_bf16out) {
        if ((N & 127) == 0) {
            dim3 grid((unsigned)(N / 128), (unsigned)(BATCH / 128));
            if (elu_bf16out)
                gemm_bf16_wmma<128, 1, 1><<<grid, 256, 0, stream>>>(
                    Am, lda, Bw, N, bias, nullptr, (bf16*)out, N, K);
            else
                gemm_bf16_wmma<128, 0, 0><<<grid, 256, 0, stream>>>(
                    Am, lda, Bw, N, bias, (float*)out, nullptr, N, K);
        } else {
            dim3 grid((unsigned)(N / 64), (unsigned)(BATCH / 128));
            if (elu_bf16out)
                gemm_bf16_wmma<64, 1, 1><<<grid, 256, 0, stream>>>(
                    Am, lda, Bw, N, bias, nullptr, (bf16*)out, N, K);
            else
                gemm_bf16_wmma<64, 0, 0><<<grid, 256, 0, stream>>>(
                    Am, lda, Bw, N, bias, (float*)out, nullptr, N, K);
        }
    };

    const int EW = 256;
    const unsigned nDS = (BATCH * DS_DIM + EW - 1) / EW;
    const unsigned nH  = (BATCH * H_DIM  + EW - 1) / EW;
    const unsigned nDE = (BATCH * DE_DIM + EW - 1) / EW;

    for (int t = 0; t < T_STEPS; ++t) {
        step_begin<<<nDS, EW, 0, stream>>>(stateF, hB, reset, t);
        za_kernel <<<nH,  EW, 0, stream>>>(stateF, action, za_W, za_b, zaB, t);

        gemm(zaB, H_DIM, WiB, gru_bi, giF, G_DIM, H_DIM, false);   // gi
        gemm(hB,  D_DIM, WhB, gru_bh, ghF, G_DIM, D_DIM, false);   // gh

        gru_gate<<<nDE, EW, 0, stream>>>(giF, ghF, stateF, heB, embed, t);

        gemm(heB, DE_DIM, P1B, prior_b1, p1B, H_DIM,    D_DIM,  true);   // prior L1 (elu)
        gemm(p1B, H_DIM,  P2B, prior_b2, priR, 2*S_DIM, H_DIM,  false);  // prior L2
        gemm(heB, DE_DIM, Q1B, post_b1,  q1B, H_DIM,    DE_DIM, true);   // post  L1 (elu)
        gemm(q1B, H_DIM,  Q2B, post_b2,  posR, 2*S_DIM, H_DIM,  false);  // post  L2

        finalize<<<nDS, EW, 0, stream>>>(priR, posR, noise, stateF,
                                         priors, posts, states, t);
    }
}
